// SparseConvNet_3985729650742
// MI455X (gfx1250) — matrix operand order repaired and checked by reading, not compiled, not run on hardware
//
#include <hip/hip_runtime.h>

// ---------- types ----------
typedef __attribute__((ext_vector_type(16))) __bf16 v16bf;
typedef __attribute__((ext_vector_type(8)))  __bf16 v8bf;
typedef __attribute__((ext_vector_type(8)))  float  v8f;
typedef unsigned short bf16_t;   // raw bf16 storage

#define EPSV 1e-4f

union ABf { v16bf v; v8bf h[2]; unsigned u[8]; };

__device__ __forceinline__ bf16_t f2bf(float f) {
    union { float f; unsigned u; } x; x.f = f;
    unsigned r = x.u + 0x7FFFu + ((x.u >> 16) & 1u);   // round-to-nearest-even
    return (bf16_t)(r >> 16);
}

__device__ __forceinline__ v8f wmma_bf16(v16bf a, v16bf b, v8f c) {
    // D = A(16x32 bf16) * B(32x16 bf16) + C(16x16 f32)
    return __builtin_amdgcn_wmma_f32_16x16x32_bf16(false, a, false, b, (short)0, c, false, false);
}

// ---------------------------------------------------------------------------
// 27-tap submanifold conv: out[N x Cout] = sum_k gather(x, nbr[:,k]) @ W[k]
// x: bf16 [N x ldx], wt: bf16 [27][Cout][Cin], out: f32 [N x ldo]
// 1 wave -> one 16x32 output tile (A gather reused for two B columns).
// ---------------------------------------------------------------------------
__global__ void smconv_wmma(const bf16_t* __restrict__ xbf, int ldx,
                            const int* __restrict__ nbr,
                            const bf16_t* __restrict__ wt,
                            float* __restrict__ out, int ldo,
                            int N, int Cin, int Cout)
{
    const int lane    = threadIdx.x & 31;
    const int wv      = threadIdx.x >> 5;
    const int rowBase = (blockIdx.x * 4 + wv) * 16;
    if (rowBase >= N) return;                      // wave-uniform
    const int coutBase = blockIdx.y * 32;
    const int m     = lane & 15;
    const int row   = rowBase + m;
    const bool rv   = row < N;
    const int kbase = (lane & 16) ? 16 : 0;        // A-frag K offset (contiguous k = 16h + s)
    const int bsel  = (lane & 16) ? 16 : 0;        // B-frag K offset (contiguous k = 16h + s, same as A)
    const int col0  = coutBase + m;
    const int col1  = col0 + 16;
    const int kcn   = Cin >> 5;

    v8f acc0 = {}, acc1 = {};
    for (int k = 0; k < 27; ++k) {
        int idx = rv ? nbr[row * 27 + k] : -1;
        const bf16_t* ar  = xbf + (long)(idx < 0 ? 0 : idx) * ldx + kbase;
        const bf16_t* br0 = wt + ((long)k * Cout + col0) * (long)Cin + bsel;
        const bf16_t* br1 = wt + ((long)k * Cout + col1) * (long)Cin + bsel;
        for (int kc = 0; kc < kcn; ++kc) {
            ABf a;
            a.h[0] = *(const v8bf*)(ar + kc * 32);
            a.h[1] = *(const v8bf*)(ar + kc * 32 + 8);
            if (idx < 0) {
                #pragma unroll
                for (int j = 0; j < 8; ++j) a.u[j] = 0u;   // masked neighbor -> zero A row
            }
            v16bf b0 = *(const v16bf*)(br0 + kc * 32);
            v16bf b1 = *(const v16bf*)(br1 + kc * 32);
            acc0 = wmma_bf16(a.v, b0, acc0);
            acc1 = wmma_bf16(a.v, b1, acc1);
        }
    }
    #pragma unroll
    for (int r = 0; r < 8; ++r) {
        int orow = rowBase + r + ((lane & 16) ? 8 : 0);
        if (orow < N) {
            out[(long)orow * ldo + col0] = acc0[r];
            out[(long)orow * ldo + col1] = acc1[r];
        }
    }
}

// ---------------------------------------------------------------------------
// Down conv: contrib[i] = x[i] @ W[ko[i]]; out[parent[i]] += contrib[i]
// 8 masked WMMA passes per K-chunk, then f32 atomic scatter. 16x32 tile/wave.
// ---------------------------------------------------------------------------
__global__ void downconv_wmma(const bf16_t* __restrict__ xbf, int ldx,
                              const int* __restrict__ parent,
                              const int* __restrict__ ko,
                              const bf16_t* __restrict__ wt,
                              float* __restrict__ out, int ldo,
                              int N, int Cin, int Cout)
{
    const int lane    = threadIdx.x & 31;
    const int wv      = threadIdx.x >> 5;
    const int rowBase = (blockIdx.x * 4 + wv) * 16;
    if (rowBase >= N) return;
    const int coutBase = blockIdx.y * 32;
    const int m     = lane & 15;
    const int row   = rowBase + m;
    const bool rv   = row < N;
    const int kbase = (lane & 16) ? 16 : 0;
    const int bsel  = (lane & 16) ? 16 : 0;
    const int col0  = coutBase + m;
    const int col1  = col0 + 16;
    const int kcn   = Cin >> 5;
    const int myko  = rv ? ko[row] : -1;
    const bf16_t* ar = xbf + (long)(rv ? row : 0) * ldx + kbase;

    v8f acc0 = {}, acc1 = {};
    for (int kc = 0; kc < kcn; ++kc) {
        ABf a;
        a.h[0] = *(const v8bf*)(ar + kc * 32);
        a.h[1] = *(const v8bf*)(ar + kc * 32 + 8);
        for (int k = 0; k < 8; ++k) {
            ABf am;
            bool use = (myko == k);
            #pragma unroll
            for (int j = 0; j < 8; ++j) am.u[j] = use ? a.u[j] : 0u;
            v16bf b0 = *(const v16bf*)(wt + ((long)k * Cout + col0) * (long)Cin + bsel + kc * 32);
            v16bf b1 = *(const v16bf*)(wt + ((long)k * Cout + col1) * (long)Cin + bsel + kc * 32);
            acc0 = wmma_bf16(am.v, b0, acc0);
            acc1 = wmma_bf16(am.v, b1, acc1);
        }
    }
    #pragma unroll
    for (int r = 0; r < 8; ++r) {
        int orow = rowBase + r + ((lane & 16) ? 8 : 0);
        if (orow < N) {
            long pr = (long)parent[orow] * ldo;
            atomicAdd(&out[pr + col0], acc0[r]);
            atomicAdd(&out[pr + col1], acc1[r]);
        }
    }
}

// ---------------------------------------------------------------------------
// Up conv: out[i] = xc[parent[i]] @ W[ko[i]]  (direct store). 16x32 tile/wave.
// ---------------------------------------------------------------------------
__global__ void upconv_wmma(const bf16_t* __restrict__ xbf, int ldx,
                            const int* __restrict__ parent,
                            const int* __restrict__ ko,
                            const bf16_t* __restrict__ wt,
                            float* __restrict__ out, int ldo,
                            int N, int Cin, int Cout)
{
    const int lane    = threadIdx.x & 31;
    const int wv      = threadIdx.x >> 5;
    const int rowBase = (blockIdx.x * 4 + wv) * 16;
    if (rowBase >= N) return;
    const int coutBase = blockIdx.y * 32;
    const int m     = lane & 15;
    const int row   = rowBase + m;
    const bool rv   = row < N;
    const int kbase = (lane & 16) ? 16 : 0;
    const int bsel  = (lane & 16) ? 16 : 0;
    const int col0  = coutBase + m;
    const int col1  = col0 + 16;
    const int kcn   = Cin >> 5;
    const int myko  = rv ? ko[row] : -1;
    const int gidx  = rv ? parent[row] : 0;
    const bf16_t* ar = xbf + (long)gidx * ldx + kbase;

    v8f acc0 = {}, acc1 = {};
    for (int kc = 0; kc < kcn; ++kc) {
        ABf a;
        a.h[0] = *(const v8bf*)(ar + kc * 32);
        a.h[1] = *(const v8bf*)(ar + kc * 32 + 8);
        for (int k = 0; k < 8; ++k) {
            ABf am;
            bool use = (myko == k);
            #pragma unroll
            for (int j = 0; j < 8; ++j) am.u[j] = use ? a.u[j] : 0u;
            v16bf b0 = *(const v16bf*)(wt + ((long)k * Cout + col0) * (long)Cin + bsel + kc * 32);
            v16bf b1 = *(const v16bf*)(wt + ((long)k * Cout + col1) * (long)Cin + bsel + kc * 32);
            acc0 = wmma_bf16(am.v, b0, acc0);
            acc1 = wmma_bf16(am.v, b1, acc1);
        }
    }
    #pragma unroll
    for (int r = 0; r < 8; ++r) {
        int orow = rowBase + r + ((lane & 16) ? 8 : 0);
        if (orow < N) {
            out[(long)orow * ldo + col0] = acc0[r];
            out[(long)orow * ldo + col1] = acc1[r];
        }
    }
}

// ---------------------------------------------------------------------------
// BatchNorm statistics: stats[0:C]=sum, stats[C:2C]=sumsq (pre-zeroed).
// Each block reduces 256 rows; register partials, one atomic per channel.
// ---------------------------------------------------------------------------
__global__ void bn_stats(const float* __restrict__ x, int ld, int N, int C,
                         float* __restrict__ stats)
{
    float s0 = 0.f, s1 = 0.f, q0 = 0.f, q1 = 0.f;
    int c0 = threadIdx.x, c1 = threadIdx.x + 256;
    int r0 = blockIdx.x * 256;
    int r1 = r0 + 256; if (r1 > N) r1 = N;
    for (int i = r0; i < r1; ++i) {
        const float* xr = x + (long)i * ld;
        if (c0 < C) { float v = xr[c0]; s0 += v; q0 += v * v; }
        if (c1 < C) { float v = xr[c1]; s1 += v; q1 += v * v; }
    }
    if (c0 < C) { atomicAdd(&stats[c0], s0); atomicAdd(&stats[C + c0], q0); }
    if (c1 < C) { atomicAdd(&stats[c1], s1); atomicAdd(&stats[C + c1], q1); }
}

// bn+relu apply: optional bf16 output (next conv input) and/or f32 output.
__global__ void bn_apply(const float* __restrict__ x, int ldx, int N, int C,
                         const float* __restrict__ stats, float invN,
                         const float* __restrict__ g, const float* __restrict__ b,
                         bf16_t* __restrict__ obf, int ldob,
                         float* __restrict__ of32, int ldof)
{
    int c = blockIdx.y * blockDim.x + threadIdx.x;
    if (c >= C) return;
    float mu  = stats[c] * invN;
    float var = fmaf(-mu, mu, stats[C + c] * invN);
    float sc  = rsqrtf(var + EPSV) * g[c];
    float sh  = b[c];
    int r0 = blockIdx.x * 4;
    int r1 = r0 + 4; if (r1 > N) r1 = N;
    for (int i = r0; i < r1; ++i) {
        float y = fmaxf(fmaf(x[(long)i * ldx + c] - mu, sc, sh), 0.f);
        if (obf)  obf[(long)i * ldob + c]  = f2bf(y);
        if (of32) of32[(long)i * ldof + c] = y;
    }
}

// Transpose + convert weights: w f32 [K][Ci][Co] -> wt bf16 [K][Co][Cit] (Ci padded to Cit).
__global__ void conv_weight(const float* __restrict__ w, bf16_t* __restrict__ wt,
                            int K, int Ci, int Co, int Cit, long total)
{
    for (long i = (long)blockIdx.x * blockDim.x + threadIdx.x; i < total;
         i += (long)gridDim.x * blockDim.x) {
        int  ci = (int)(i % Cit);
        long t  = i / Cit;
        int  co = (int)(t % Co);
        int  k  = (int)(t / Co);
        float v = (ci < Ci) ? w[((long)k * Ci + ci) * Co + co] : 0.f;
        wt[i] = f2bf(v);
    }
}

// feats f32 [N x 3] -> bf16 [N x 32] zero-padded (so input conv uses same WMMA path)
__global__ void pad_feats(const float* __restrict__ f, bf16_t* __restrict__ o, long total)
{
    for (long i = (long)blockIdx.x * blockDim.x + threadIdx.x; i < total;
         i += (long)gridDim.x * blockDim.x) {
        int  c = (int)(i & 31);
        long r = i >> 5;
        o[i] = f2bf(c < 3 ? f[r * 3 + c] : 0.f);
    }
}

// ---------------------------------------------------------------------------
extern "C" void kernel_launch(void* const* d_in, const int* in_sizes, int n_in,
                              void* d_out, int out_size, void* d_ws, size_t ws_size,
                              hipStream_t stream)
{
    (void)n_in; (void)out_size; (void)ws_size;
    // input order = setup_inputs() dict order:
    // nbr0,parent0,ko0, nbr1,parent1,ko1, ..., nbr5,parent5,ko5, nbr6, params, feats
    const int* nbr[7]; const int* parent[6]; const int* ko[6];
    for (int l = 0; l < 6; ++l) {
        nbr[l]    = (const int*)d_in[3 * l];
        parent[l] = (const int*)d_in[3 * l + 1];
        ko[l]     = (const int*)d_in[3 * l + 2];
    }
    nbr[6] = (const int*)d_in[18];
    const float* params = (const float*)d_in[19];
    const float* feats  = (const float*)d_in[20];

    int Nl[7];
    for (int l = 0; l < 6; ++l) Nl[l] = in_sizes[3 * l] / 27;
    Nl[6] = in_sizes[18] / 27;
    int P[8]; for (int l = 0; l < 8; ++l) P[l] = 32 * (l + 1);

    // ---- parameter offsets (param_shapes order) ----
    long poff = 0;
    auto take = [&](long n) { long o = poff; poff += n; return o; };
    long o_Win = take(27L * 3 * 32);
    long o_gpre[7], o_bpre[7], o_Wpre[7];
    for (int l = 0; l < 7; ++l) {
        o_gpre[l] = take(P[l]); o_bpre[l] = take(P[l]); o_Wpre[l] = take(27L * P[l] * P[l]);
    }
    long o_gdn[6], o_bdn[6], o_Wdn[6], o_gup[6], o_bup[6], o_Wup[6], o_gpo[6], o_bpo[6], o_Wpo[6];
    for (int l = 0; l < 6; ++l) {
        o_gdn[l] = take(P[l]);      o_bdn[l] = take(P[l]);      o_Wdn[l] = take(8L * P[l] * P[l + 1]);
        o_gup[l] = take(P[l + 1]);  o_bup[l] = take(P[l + 1]);  o_Wup[l] = take(8L * P[l + 1] * P[l]);
        o_gpo[l] = take(2L * P[l]); o_bpo[l] = take(2L * P[l]); o_Wpo[l] = take(27L * 2 * P[l] * P[l]);
    }
    long o_gout = take(32), o_bout = take(32);

    // ---- workspace carve-up (256B aligned bump allocator) ----
    char* base = (char*)d_ws;
    size_t woff = 0;
    auto walloc = [&](size_t bytes) -> void* {
        void* p = base + woff;
        woff = (woff + bytes + 255) & ~(size_t)255;
        return p;
    };
    bf16_t* wt_in = (bf16_t*)walloc(27L * 32 * 32 * 2);
    bf16_t *wt_pre[7], *wt_dn[6], *wt_up[6], *wt_po[6];
    for (int l = 0; l < 7; ++l) wt_pre[l] = (bf16_t*)walloc(27L * P[l] * P[l] * 2);
    for (int l = 0; l < 6; ++l) {
        wt_dn[l] = (bf16_t*)walloc(8L * P[l] * P[l + 1] * 2);
        wt_up[l] = (bf16_t*)walloc(8L * P[l + 1] * P[l] * 2);
        wt_po[l] = (bf16_t*)walloc(27L * 2 * P[l] * P[l] * 2);
    }
    bf16_t* featsbf = (bf16_t*)walloc((long)Nl[0] * 32 * 2);
    float* xin[7];                                 // level input; reused for level output
    for (int l = 0; l < 7; ++l) xin[l] = (float*)walloc((long)Nl[l] * P[l] * 4);
    float* cat[6];                                 // [x_pre | y_up] concat buffer, ld = 2P
    for (int l = 0; l < 6; ++l) cat[l] = (float*)walloc((long)Nl[l] * 2 * P[l] * 4);
    long bfmax = (long)Nl[6] * P[6];
    for (int l = 0; l < 6; ++l) { long s = (long)Nl[l] * 2 * P[l]; if (s > bfmax) bfmax = s; }
    bf16_t* bfbuf = (bf16_t*)walloc(bfmax * 2);    // shared bn->conv bf16 staging
    float* stats  = (float*)walloc(2 * 512 * 4);

    // ---- launch helpers ----
    auto launch_smconv = [&](const bf16_t* xbf, int ldx, const int* nb, const bf16_t* wt,
                             float* out, int ldo, int N, int Cin, int Cout) {
        dim3 g((unsigned)((N + 63) / 64), (unsigned)(Cout / 32));
        smconv_wmma<<<g, dim3(128), 0, stream>>>(xbf, ldx, nb, wt, out, ldo, N, Cin, Cout);
    };
    auto launch_bn = [&](const float* x, int ldx, int N, int C, long og, long ob,
                         bf16_t* obf, int ldob, float* of32, int ldof) {
        hipMemsetAsync(stats, 0, (size_t)2 * C * sizeof(float), stream);
        bn_stats<<<dim3((unsigned)((N + 255) / 256)), dim3(256), 0, stream>>>(x, ldx, N, C, stats);
        dim3 ga((unsigned)((N + 3) / 4), (unsigned)((C + 255) / 256));
        bn_apply<<<ga, dim3(256), 0, stream>>>(x, ldx, N, C, stats, 1.0f / (float)N,
                                               params + og, params + ob, obf, ldob, of32, ldof);
    };
    auto cw = [&](long ow, bf16_t* wt, int K, int Ci, int Co, int Cit) {
        long total = (long)K * Co * Cit;
        long blocks = (total + 255) / 256; if (blocks > 4096) blocks = 4096;
        conv_weight<<<dim3((unsigned)blocks), dim3(256), 0, stream>>>(params + ow, wt, K, Ci, Co, Cit, total);
    };

    // ---- one-time-per-call prep: weights to bf16 [k][Co][Ci], feats padded ----
    cw(o_Win, wt_in, 27, 3, 32, 32);
    for (int l = 0; l < 7; ++l) cw(o_Wpre[l], wt_pre[l], 27, P[l], P[l], P[l]);
    for (int l = 0; l < 6; ++l) {
        cw(o_Wdn[l], wt_dn[l], 8, P[l], P[l + 1], P[l]);
        cw(o_Wup[l], wt_up[l], 8, P[l + 1], P[l], P[l + 1]);
        cw(o_Wpo[l], wt_po[l], 27, 2 * P[l], P[l], 2 * P[l]);
    }
    {
        long ftot = (long)Nl[0] * 32;
        long blocks = (ftot + 255) / 256; if (blocks > 8192) blocks = 8192;
        pad_feats<<<dim3((unsigned)blocks), dim3(256), 0, stream>>>(feats, featsbf, ftot);
    }

    // ---- input conv: xin0 = smconv(feats, nbr0, W_in) ----
    launch_smconv(featsbf, 32, nbr[0], wt_in, xin[0], 32, Nl[0], 32, 32);

    // ---- downward pass: pre conv, then down conv ----
    for (int l = 0; l <= 6; ++l) {
        float* preout = (l < 6) ? cat[l] : xin[6];
        int    ldpre  = (l < 6) ? 2 * P[l] : P[6];
        launch_bn(xin[l], P[l], Nl[l], P[l], o_gpre[l], o_bpre[l], bfbuf, P[l], nullptr, 0);
        launch_smconv(bfbuf, P[l], nbr[l], wt_pre[l], preout, ldpre, Nl[l], P[l], P[l]);
        if (l < 6) {
            launch_bn(cat[l], 2 * P[l], Nl[l], P[l], o_gdn[l], o_bdn[l], bfbuf, P[l], nullptr, 0);
            hipMemsetAsync(xin[l + 1], 0, (size_t)Nl[l + 1] * P[l + 1] * sizeof(float), stream);
            dim3 g((unsigned)((Nl[l] + 63) / 64), (unsigned)(P[l + 1] / 32));
            downconv_wmma<<<g, dim3(128), 0, stream>>>(bfbuf, P[l], parent[l], ko[l], wt_dn[l],
                                                       xin[l + 1], P[l + 1], Nl[l], P[l], P[l + 1]);
        }
    }

    // ---- upward pass: up conv into cat[:,P:2P], then post conv ----
    for (int l = 5; l >= 0; --l) {
        launch_bn(xin[l + 1], P[l + 1], Nl[l + 1], P[l + 1], o_gup[l], o_bup[l],
                  bfbuf, P[l + 1], nullptr, 0);
        dim3 g((unsigned)((Nl[l] + 63) / 64), (unsigned)(P[l] / 32));
        upconv_wmma<<<g, dim3(128), 0, stream>>>(bfbuf, P[l + 1], parent[l], ko[l], wt_up[l],
                                                 cat[l] + P[l], 2 * P[l], Nl[l], P[l + 1], P[l]);
        launch_bn(cat[l], 2 * P[l], Nl[l], 2 * P[l], o_gpo[l], o_bpo[l], bfbuf, 2 * P[l], nullptr, 0);
        launch_smconv(bfbuf, 2 * P[l], nbr[l], wt_po[l], xin[l], P[l], Nl[l], 2 * P[l], P[l]);
    }

    // ---- final bn+relu -> d_out (f32, N0 x 32) ----
    launch_bn(xin[0], 32, Nl[0], 32, o_gout, o_bout, nullptr, 0, (float*)d_out, 32);
}
